// GriffinBlock_62792421868024
// MI455X (gfx1250) — matrix operand order repaired and checked by reading, not compiled, not run on hardware
//
#include <hip/hip_runtime.h>
#include <math.h>

// ---------------------------------------------------------------------------
// Griffin block (RG-LRU mixer + SwiGLU FFN) for gfx1250 (CDNA5, wave32).
// GEMMs run on V_WMMA_F32_16X16X32_BF16 using a bf16x3 split
// (a_hi*b_hi + a_hi*b_lo + a_lo*b_hi, f32 accumulate) for fp32-class accuracy
// at ~2.7x the matrix throughput of the fp32 16x16x4 WMMA path.
// ---------------------------------------------------------------------------

typedef __attribute__((ext_vector_type(8)))  __bf16 v8bf;
typedef __attribute__((ext_vector_type(16))) __bf16 v16bf;
typedef __attribute__((ext_vector_type(8)))  float  v8f;

constexpr int BATCH  = 4;
constexpr int SEQ    = 2048;
constexpr int DMODEL = 2048;
constexpr int DFF    = 4096;
constexpr int MROWS  = BATCH * SEQ;          // 8192
constexpr float RMS_EPS = 1e-6f;
constexpr float LRU_EPS = 1e-6f;
constexpr int CHUNK = 128;                   // scan chunk length
constexpr int NCHUNK = SEQ / CHUNK;          // 16

__device__ __forceinline__ float sigmoidf_(float x) {
  return 1.0f / (1.0f + __expf(-x));
}
__device__ __forceinline__ float alpha_of(float ld) {
  float a = sigmoidf_(ld);
  return fminf(fmaxf(a, LRU_EPS), 1.0f - LRU_EPS);
}
__device__ __forceinline__ void split_bf16(float v, __bf16& h, __bf16& l) {
  h = (__bf16)v;
  l = (__bf16)(v - (float)h);
}

// A fragment (16x32 bf16, ISA 7.12.2): lane half 0 holds K=0..7 & 16..23,
// half 1 holds K=8..15 & 24..31 -> two 16B chunks from the LDS row.
__device__ __forceinline__ v16bf load_frag_a(const __bf16* row, int hsel) {
  v8bf x0 = *(const v8bf*)(row + hsel * 8);
  v8bf x1 = *(const v8bf*)(row + 16 + hsel * 8);
  return __builtin_shufflevector(x0, x1, 0, 1, 2, 3, 4, 5, 6, 7,
                                 8, 9, 10, 11, 12, 13, 14, 15);
}
// B fragment (32x16 bf16): lane half 0 holds K=0..7 & 16..23, half 1 holds
// K=8..15 & 24..31 of column n (same K order as the A fragment).
__device__ __forceinline__ v16bf load_frag_b(const __bf16* row, int hsel) {
  v8bf x0 = *(const v8bf*)(row + hsel * 8);
  v8bf x1 = *(const v8bf*)(row + 16 + hsel * 8);
  return __builtin_shufflevector(x0, x1, 0, 1, 2, 3, 4, 5, 6, 7,
                                 8, 9, 10, 11, 12, 13, 14, 15);
}

// ---------------------------------------------------------------------------
// bf16x3 WMMA GEMM:  out[M,N] = A[M,K] @ W[N,K]^T  (+ epilogue)
//   A, W supplied as hi/lo bf16 pairs (row-major, K contiguous).
// MODE 0: outF = acc
// MODE 1: outF = sigmoid(acc + extra[n])           (bias vector)
// MODE 2: outF = extra[m,n] + acc                  (residual add)
// MODE 3: outF = silu(acc)
// MODE 4: split(acc * extra[m,n]) -> outH/outL     (gated, bf16-split output)
// Block: 256 threads = 8 waves (4x2). Tile BM=128 x BN=128, BK=32.
// Each wave: 32x64 C tile = 2x4 fragments; 24 bf16 WMMAs per BK step.
// ---------------------------------------------------------------------------
template <int MODE>
__global__ __launch_bounds__(256)
void gemm_bf16x3(const __bf16* __restrict__ Ahi, const __bf16* __restrict__ Alo,
                 const __bf16* __restrict__ Whi, const __bf16* __restrict__ Wlo,
                 float* __restrict__ outF, __bf16* __restrict__ outH,
                 __bf16* __restrict__ outL, const float* __restrict__ extra,
                 int N, int K) {
  constexpr int BM = 128, BN = 128, BK = 32, LK = BK + 8;  // +8 bf16 pad
  __shared__ alignas(16) __bf16 AhS[BM][LK];
  __shared__ alignas(16) __bf16 AlS[BM][LK];
  __shared__ alignas(16) __bf16 WhS[BN][LK];
  __shared__ alignas(16) __bf16 WlS[BN][LK];

  const int tid  = threadIdx.x;
  const int wave = tid >> 5;
  const int lane = tid & 31;
  const int hsel = lane >> 4;          // 0: lanes 0-15, 1: lanes 16-31
  const int l16  = lane & 15;
  const int mTile = blockIdx.y * BM;
  const int nTile = blockIdx.x * BN;
  const int wm = (wave >> 1) * 32;     // 0,32,64,96
  const int wn = (wave & 1) * 64;      // 0,64

  const int lr  = tid >> 2;            // 0..63 : row loaded by this thread
  const int lc8 = (tid & 3) * 8;       // 0,8,16,24 : k column (8 bf16 = 16B)

  v8f c[2][4] = {};

  for (int k0 = 0; k0 < K; k0 += BK) {
    // stage hi/lo tiles through registers (two 64-row passes per tile)
    v8bf rAh[2], rAl[2], rWh[2], rWl[2];
#pragma unroll
    for (int p = 0; p < 2; ++p) {
      const size_t ga = (size_t)(mTile + p * 64 + lr) * K + k0 + lc8;
      const size_t gw = (size_t)(nTile + p * 64 + lr) * K + k0 + lc8;
      rAh[p] = *(const v8bf*)(Ahi + ga);
      rAl[p] = *(const v8bf*)(Alo + ga);
      rWh[p] = *(const v8bf*)(Whi + gw);
      rWl[p] = *(const v8bf*)(Wlo + gw);
    }
    if (k0 + BK < K) {  // global_prefetch_b8 of next K-tile
      __builtin_prefetch(Ahi + (size_t)(mTile + lr) * K + (k0 + BK) + lc8, 0, 1);
      __builtin_prefetch(Whi + (size_t)(nTile + lr) * K + (k0 + BK) + lc8, 0, 1);
    }
    __syncthreads();
#pragma unroll
    for (int p = 0; p < 2; ++p) {
      *(v8bf*)&AhS[p * 64 + lr][lc8] = rAh[p];
      *(v8bf*)&AlS[p * 64 + lr][lc8] = rAl[p];
      *(v8bf*)&WhS[p * 64 + lr][lc8] = rWh[p];
      *(v8bf*)&WlS[p * 64 + lr][lc8] = rWl[p];
    }
    __syncthreads();

    v16bf ah[2], al[2];
#pragma unroll
    for (int i = 0; i < 2; ++i) {
      const int m = wm + i * 16 + l16;
      ah[i] = load_frag_a(&AhS[m][0], hsel);
      al[i] = load_frag_a(&AlS[m][0], hsel);
    }
#pragma unroll
    for (int j = 0; j < 4; ++j) {
      const int n = wn + j * 16 + l16;
      const v16bf bh = load_frag_b(&WhS[n][0], hsel);
      const v16bf bl = load_frag_b(&WlS[n][0], hsel);
#pragma unroll
      for (int i = 0; i < 2; ++i) {
        c[i][j] = __builtin_amdgcn_wmma_f32_16x16x32_bf16(
            false, ah[i], false, bh, (short)0, c[i][j], false, false);
        c[i][j] = __builtin_amdgcn_wmma_f32_16x16x32_bf16(
            false, ah[i], false, bl, (short)0, c[i][j], false, false);
        c[i][j] = __builtin_amdgcn_wmma_f32_16x16x32_bf16(
            false, al[i], false, bh, (short)0, c[i][j], false, false);
      }
    }
  }

  // epilogue: C/D layout: VGPR e -> M = e + 8*hsel ; N = l16
#pragma unroll
  for (int i = 0; i < 2; ++i)
#pragma unroll
    for (int j = 0; j < 4; ++j) {
      const int nIdx = nTile + wn + j * 16 + l16;
#pragma unroll
      for (int e = 0; e < 8; ++e) {
        const int mIdx = mTile + wm + i * 16 + hsel * 8 + e;
        const size_t idx = (size_t)mIdx * N + nIdx;
        const float v = c[i][j][e];
        if (MODE == 0)      outF[idx] = v;
        else if (MODE == 1) outF[idx] = sigmoidf_(v + extra[nIdx]);
        else if (MODE == 2) outF[idx] = extra[idx] + v;
        else if (MODE == 3) outF[idx] = v * sigmoidf_(v);
        else {  // MODE 4: gated multiply, bf16-split output for next GEMM
          const float u = v * extra[idx];
          __bf16 h, l; split_bf16(u, h, l);
          outH[idx] = h; outL[idx] = l;
        }
      }
    }
}

// ---------------------------------------------------------------------------
// f32 -> bf16 hi/lo split (weights; once per launch).
// ---------------------------------------------------------------------------
__global__ __launch_bounds__(256)
void cvt_split_kernel(const float* __restrict__ in, __bf16* __restrict__ hi,
                      __bf16* __restrict__ lo) {
  const size_t i = (size_t)blockIdx.x * 256 + threadIdx.x;
  __bf16 h, l; split_bf16(in[i], h, l);
  hi[i] = h; lo[i] = l;
}

// ---------------------------------------------------------------------------
// RMSNorm with fused bf16 hi/lo split output.  One block per row of D=2048.
// ---------------------------------------------------------------------------
__global__ __launch_bounds__(256)
void rmsnorm_split_kernel(const float* __restrict__ x, const float* __restrict__ w,
                          __bf16* __restrict__ nhi, __bf16* __restrict__ nlo) {
  __shared__ float red[256];
  const size_t row = blockIdx.x;
  const float* xr = x + row * DMODEL;
  float s = 0.0f;
#pragma unroll
  for (int i = 0; i < DMODEL / 256; ++i) {
    const float v = xr[threadIdx.x + i * 256];
    s += v * v;
  }
  red[threadIdx.x] = s;
  __syncthreads();
  for (int off = 128; off > 0; off >>= 1) {
    if (threadIdx.x < off) red[threadIdx.x] += red[threadIdx.x + off];
    __syncthreads();
  }
  const float inv = rsqrtf(red[0] / (float)DMODEL + RMS_EPS);
#pragma unroll
  for (int i = 0; i < DMODEL / 256; ++i) {
    const int col = threadIdx.x + i * 256;
    __bf16 h, l; split_bf16(xr[col] * inv * w[col], h, l);
    nhi[row * DMODEL + col] = h;
    nlo[row * DMODEL + col] = l;
  }
}

// ---------------------------------------------------------------------------
// RG-LRU scan, chunk-parallel (3 passes).
// ---------------------------------------------------------------------------
__global__ __launch_bounds__(256)
void scan_chunk_kernel(const float* __restrict__ r, const float* __restrict__ xp,
                       const float* __restrict__ log_decay,
                       float* __restrict__ hs, float* __restrict__ partial) {
  const int idx = blockIdx.x * 256 + threadIdx.x;   // B*NCHUNK*D threads
  const int d  = idx % DMODEL;
  const int bc = idx / DMODEL;
  const int c  = bc % NCHUNK;
  const int b  = bc / NCHUNK;
  const float alpha = alpha_of(log_decay[d]);
  const float scale = sqrtf(fmaxf(1.0f - alpha * alpha, 0.0f));
  float h = 0.0f;
  const size_t base = ((size_t)b * SEQ + (size_t)c * CHUNK) * DMODEL + d;
  for (int i = 0; i < CHUNK; ++i) {
    const size_t p = base + (size_t)i * DMODEL;
    h = alpha * h + scale * (r[p] * xp[p]);
    hs[p] = h;
  }
  partial[idx] = h;   // layout (b, c, d)
}

__global__ __launch_bounds__(256)
void scan_prefix_kernel(const float* __restrict__ state,
                        const float* __restrict__ log_decay,
                        const float* __restrict__ partial,
                        float* __restrict__ pre, float* __restrict__ h_final) {
  const int idx = blockIdx.x * 256 + threadIdx.x;   // B*D threads
  const int d = idx % DMODEL;
  const int b = idx / DMODEL;
  const float alpha = alpha_of(log_decay[d]);
  const float aCH = __powf(alpha, (float)CHUNK);
  float p = state[idx];
  for (int c = 0; c < NCHUNK; ++c) {
    const size_t o = ((size_t)b * NCHUNK + c) * DMODEL + d;
    pre[o] = p;
    p = aCH * p + partial[o];
  }
  h_final[idx] = p;
}

// Pass 3: h = hs_local + alpha^(tl+1)*pre; fused output gate *g; bf16 split out.
__global__ __launch_bounds__(256)
void scan_fix_kernel(const float* __restrict__ hs, const float* __restrict__ pre,
                     const float* __restrict__ g,
                     const float* __restrict__ log_decay,
                     __bf16* __restrict__ yhi, __bf16* __restrict__ ylo) {
  const size_t idx = (size_t)blockIdx.x * 256 + threadIdx.x;  // B*T*D threads
  const int d = (int)(idx % DMODEL);
  const size_t bt = idx / DMODEL;
  const int t = (int)(bt % SEQ);
  const int b = (int)(bt / SEQ);
  const int c = t / CHUNK, tl = t % CHUNK;
  const float alpha = alpha_of(log_decay[d]);
  const float corr = __powf(alpha, (float)(tl + 1));
  const float h = hs[idx] + corr * pre[((size_t)b * NCHUNK + c) * DMODEL + d];
  __bf16 hh, ll; split_bf16(h * g[idx], hh, ll);
  yhi[idx] = hh; ylo[idx] = ll;
}

// ---------------------------------------------------------------------------
extern "C" void kernel_launch(void* const* d_in, const int* in_sizes, int n_in,
                              void* d_out, int out_size, void* d_ws, size_t ws_size,
                              hipStream_t stream) {
  const float* x         = (const float*)d_in[0];
  const float* state     = (const float*)d_in[1];
  const float* w_norm1   = (const float*)d_in[2];
  const float* W_in      = (const float*)d_in[3];
  const float* W_r       = (const float*)d_in[4];
  const float* b_r       = (const float*)d_in[5];
  const float* W_i       = (const float*)d_in[6];
  const float* b_i       = (const float*)d_in[7];
  const float* log_decay = (const float*)d_in[8];
  const float* W_out     = (const float*)d_in[9];
  const float* w_norm2   = (const float*)d_in[10];
  const float* W_gate    = (const float*)d_in[11];
  const float* W_up      = (const float*)d_in[12];
  const float* W_down    = (const float*)d_in[13];
  (void)in_sizes; (void)n_in; (void)out_size; (void)ws_size;

  float* out     = (float*)d_out;                  // (B,T,D) flattened
  float* h_final = out + (size_t)MROWS * DMODEL;   // then (B,D)

  const size_t SZ  = (size_t)MROWS * DMODEL;       // 16.78M
  const size_t SZF = (size_t)MROWS * DFF;          // 33.55M (== 2*SZ)
  const size_t WD  = (size_t)DMODEL * DMODEL;      // 4.19M
  const size_t WF  = (size_t)DFF * DMODEL;         // 8.39M

  float* ws   = (float*)d_ws;
  float* rbuf = ws;               // r        | later: ffg (silu gate, SZF f32)
  float* xpb  = ws + 1 * SZ;      // xp       | (second half of ffg)
  float* gbuf = ws + 2 * SZ;      // g        | later: uH (SZF bf16)
  float* hsb  = ws + 3 * SZ;      // hs       | later: uL (SZF bf16)
  float* x1   = ws + 4 * SZ;      // mixer residual (lives to the end)
  __bf16* nH  = (__bf16*)(ws + 5 * SZ);            // n / n2 hi   (SZ bf16)
  __bf16* nL  = nH + SZ;                           // n / n2 lo
  __bf16* yH  = (__bf16*)(ws + 6 * SZ);            // y hi        (SZ bf16)
  __bf16* yL  = yH + SZ;                           // y lo
  __bf16* wbf = (__bf16*)(ws + 7 * SZ);            // weight hi/lo pool
  __bf16* WrH = wbf;            __bf16* WrL = WrH + WD;
  __bf16* WiH = WrL + WD;       __bf16* WiL = WiH + WD;
  __bf16* WnH = WiL + WD;       __bf16* WnL = WnH + WD;   // W_in
  __bf16* WoH = WnL + WD;       __bf16* WoL = WoH + WD;   // W_out
  __bf16* WgH = WoL + WD;       __bf16* WgL = WgH + WF;
  __bf16* WuH = WgL + WF;       __bf16* WuL = WuH + WF;
  __bf16* WdH = WuL + WF;       __bf16* WdL = WdH + WF;
  float* partial = (float*)(WdL + WF);             // B*NCHUNK*D
  float* pre     = partial + (size_t)BATCH * NCHUNK * DMODEL;

  const dim3 blk(256);
  const dim3 gD(DMODEL / 128, MROWS / 128);   // N=2048 GEMMs
  const dim3 gF(DFF / 128,    MROWS / 128);   // N=4096 GEMMs

  // ---- weight hi/lo split (once per launch) ----
  cvt_split_kernel<<<(unsigned)(WD / 256), blk, 0, stream>>>(W_r,    WrH, WrL);
  cvt_split_kernel<<<(unsigned)(WD / 256), blk, 0, stream>>>(W_i,    WiH, WiL);
  cvt_split_kernel<<<(unsigned)(WD / 256), blk, 0, stream>>>(W_in,   WnH, WnL);
  cvt_split_kernel<<<(unsigned)(WD / 256), blk, 0, stream>>>(W_out,  WoH, WoL);
  cvt_split_kernel<<<(unsigned)(WF / 256), blk, 0, stream>>>(W_gate, WgH, WgL);
  cvt_split_kernel<<<(unsigned)(WF / 256), blk, 0, stream>>>(W_up,   WuH, WuL);
  cvt_split_kernel<<<(unsigned)(WF / 256), blk, 0, stream>>>(W_down, WdH, WdL);

  // ---- pre-norm + mixer projections ----
  rmsnorm_split_kernel<<<MROWS, blk, 0, stream>>>(x, w_norm1, nH, nL);
  gemm_bf16x3<1><<<gD, blk, 0, stream>>>(nH, nL, WrH, WrL, rbuf, nullptr, nullptr,
                                         b_r, DMODEL, DMODEL);   // r = sigmoid
  gemm_bf16x3<1><<<gD, blk, 0, stream>>>(nH, nL, WiH, WiL, gbuf, nullptr, nullptr,
                                         b_i, DMODEL, DMODEL);   // g = sigmoid
  gemm_bf16x3<0><<<gD, blk, 0, stream>>>(nH, nL, WnH, WnL, xpb, nullptr, nullptr,
                                         nullptr, DMODEL, DMODEL);

  // ---- RG-LRU scan (chunk-parallel) ----
  scan_chunk_kernel<<<(BATCH * NCHUNK * DMODEL) / 256, blk, 0, stream>>>(
      rbuf, xpb, log_decay, hsb, partial);
  scan_prefix_kernel<<<(BATCH * DMODEL) / 256, blk, 0, stream>>>(
      state, log_decay, partial, pre, h_final);
  scan_fix_kernel<<<(unsigned)(SZ / 256), blk, 0, stream>>>(
      hsb, pre, gbuf, log_decay, yH, yL);          // y = hs*g, bf16 split

  // ---- mixer output projection + residual ----
  gemm_bf16x3<2><<<gD, blk, 0, stream>>>(yH, yL, WoH, WoL, x1, nullptr, nullptr,
                                         x, DMODEL, DMODEL);

  // ---- SwiGLU FFN ----
  rmsnorm_split_kernel<<<MROWS, blk, 0, stream>>>(x1, w_norm2, nH, nL);  // n2
  float*  ffg = rbuf;             // 2*SZ f32 region (r/xp dead)
  __bf16* uH  = (__bf16*)gbuf;    // SZF bf16 (g dead)
  __bf16* uL  = (__bf16*)hsb;     // SZF bf16 (hs dead)
  gemm_bf16x3<3><<<gF, blk, 0, stream>>>(nH, nL, WgH, WgL, ffg, nullptr, nullptr,
                                         nullptr, DFF, DMODEL);  // silu(gate)
  gemm_bf16x3<4><<<gF, blk, 0, stream>>>(nH, nL, WuH, WuL, nullptr, uH, uL,
                                         ffg, DFF, DMODEL);      // up * silu
  gemm_bf16x3<2><<<gD, blk, 0, stream>>>(uH, uL, WdH, WdL, out, nullptr, nullptr,
                                         x1, DMODEL, DFF);       // + residual
  (void)SZF;
}
